// StructuredSpatialAttention_Seq_48052094108096
// MI455X (gfx1250) — matrix operand order repaired and checked by reading, not compiled, not run on hardware
//
#include <hip/hip_runtime.h>

// ---------------------------------------------------------------------------
// StructuredSpatialAttention for MI455X (gfx1250, wave32, WMMA, async-to-LDS)
//   D=512, H=8, HD=64, S=2048, B=4
//   0) convert x / in_proj_w / out_proj_w to f16
//   1) QKV GEMM (v_wmma_f32_16x16x32_f16); epilogue scatters q,k -> [b][h][s][hd],
//      v -> TRANSPOSED [b][h][hd][s] (value_gate folded in) so every GEMM is
//      A·B^T with K-contiguous operands (CDNA5 WMMA VGPR layouts, §7.12.2).
//   2) flash attention: one 128-thread block = 64 query rows of one (b,h).
//      Per 32-key step, K tile (4KB) + V tile (4KB) are staged ONCE PER BLOCK
//      into LDS with global_load_async_to_lds_b128 (ASYNCcnt) — 4x less K/V
//      traffic than per-wave fetches — then consumed by all 4 waves via
//      ds-loaded WMMA fragments. Online softmax via 16-lane shfl_xor
//      reductions; P goes C-layout -> LDS f16 -> A-layout for the P·V wmma.
//      The 537 MB score matrix is never materialized; HBM traffic ~= mask.
//   3) out-proj GEMM, f32 output.
// Workspace: ~44 MB. LDS per block: 4+4+4 = 12 KB.
// ---------------------------------------------------------------------------

typedef __attribute__((ext_vector_type(16))) _Float16 v16h;
typedef __attribute__((ext_vector_type(8)))  _Float16 v8h;
typedef __attribute__((ext_vector_type(8)))  float    v8f;

#define DD  512
#define HH  8
#define HD  64
#define SS  2048
#define BB  4

// ---- fragment loaders (this code's contiguous k order, shared by A and B) -----------
// A (16x32 f16, MxK): lane<16 -> row=lane,    K = 0..15
//                     lane>=16 -> row=lane-16, K = 16..31
__device__ inline v16h load_a16(const _Float16* base, int ldk) {
  const int lane = threadIdx.x & 31;
  const int row  = lane & 15;
  const int koff = (lane & 16) ? 16 : 0;
  const _Float16* p = base + (size_t)row * ldk + koff;
  v8h lo = *reinterpret_cast<const v8h*>(p);        // K = koff .. koff+7
  v8h hi = *reinterpret_cast<const v8h*>(p + 8);    // K = koff+8 .. koff+15
  v16h r;
#pragma unroll
  for (int i = 0; i < 8; ++i) { r[i] = lo[i]; r[8 + i] = hi[i]; }
  return r;
}

// B (32x16 f16, KxN) from B^T stored row-major (N rows, K contiguous):
// lane L -> column N = L%16; K = (L<16 ? 0..15 : 16..31), 32B contiguous.
__device__ inline v16h load_b16(const _Float16* baseT, int ldk) {
  const int lane = threadIdx.x & 31;
  const int n = lane & 15;
  const int k = (lane & 16) ? 16 : 0;
  return *reinterpret_cast<const v16h*>(baseT + (size_t)n * ldk + k);
}

__device__ inline v8f wmma16(v16h a, v16h b, v8f c) {
  return __builtin_amdgcn_wmma_f32_16x16x32_f16(false, a, false, b,
                                                (short)0, c, false, false);
}

__device__ inline float row_max16(float v) {
  v = fmaxf(v, __shfl_xor(v, 1, 32));
  v = fmaxf(v, __shfl_xor(v, 2, 32));
  v = fmaxf(v, __shfl_xor(v, 4, 32));
  v = fmaxf(v, __shfl_xor(v, 8, 32));
  return v;
}
__device__ inline float row_sum16(float v) {
  v += __shfl_xor(v, 1, 32);
  v += __shfl_xor(v, 2, 32);
  v += __shfl_xor(v, 4, 32);
  v += __shfl_xor(v, 8, 32);
  return v;
}

// async 16B copy global -> LDS (gfx1250, ASYNCcnt-tracked)
__device__ inline void async_b128(void* lds, const void* gptr) {
  unsigned loff = (unsigned)(unsigned long long)lds;   // low 32 bits of the pointer = LDS offset
  unsigned long long ga = (unsigned long long)gptr;
  asm volatile("global_load_async_to_lds_b128 %0, %1, off"
               :: "v"(loff), "v"(ga) : "memory");
}

// 16(M) x 64(N) tile, K-loop in steps of 32; A row-major MxK, B^T row-major NxK
__device__ inline void gemm_16x64(const _Float16* A, int lda,
                                  const _Float16* Bt, int ldb,
                                  int K, v8f acc[4]) {
  for (int kb = 0; kb < K; kb += 32) {
    v16h a = load_a16(A + kb, lda);
#pragma unroll
    for (int j = 0; j < 4; ++j) {
      v16h b = load_b16(Bt + (size_t)(j * 16) * ldb + kb, ldb);
      acc[j] = wmma16(a, b, acc[j]);
    }
  }
}

// ---------------------------------------------------------------------------
__global__ __launch_bounds__(256) void cvt_f32_to_f16(const float* __restrict__ in,
                                                      _Float16* __restrict__ out,
                                                      int n) {
  int i = blockIdx.x * 256 + threadIdx.x;
  if (i < n) out[i] = (_Float16)in[i];
}

// QKV projection: [8192 x 512] @ [1536 x 512]^T; scatter epilogue.
__global__ __launch_bounds__(128) void qkv_proj_kernel(
    const _Float16* __restrict__ xh, const _Float16* __restrict__ wh,
    const float* __restrict__ bias, const float* __restrict__ gate,
    _Float16* __restrict__ q, _Float16* __restrict__ k, _Float16* __restrict__ vT) {
  const int wave = threadIdx.x >> 5, lane = threadIdx.x & 31;
  const int m0 = blockIdx.x * 64 + wave * 16;
  const int n0 = blockIdx.y * 64;
  v8f acc[4] = {};
  gemm_16x64(xh + (size_t)m0 * DD, DD, wh + (size_t)n0 * DD, DD, DD, acc);

  const int half = lane >> 4, nloc = lane & 15;
#pragma unroll
  for (int j = 0; j < 4; ++j) {
    const int n = n0 + j * 16 + nloc;
    const float bn = bias[n];
#pragma unroll
    for (int r = 0; r < 8; ++r) {
      const int m = m0 + r + 8 * half;
      const int s = m >> 2, b = m & 3;           // row = s*B + b
      const float a = acc[j][r];
      if (n < DD) {                               // Q
        const int h = n >> 6, hd = n & 63;
        q[(((size_t)(b * HH + h)) * SS + s) * HD + hd] = (_Float16)(a + bn);
      } else if (n < 2 * DD) {                    // K
        const int nn = n - DD, h = nn >> 6, hd = nn & 63;
        k[(((size_t)(b * HH + h)) * SS + s) * HD + hd] = (_Float16)(a + bn);
      } else {                                    // V (gated, transposed [hd][s])
        const int nn = n - 2 * DD, h = nn >> 6, hd = nn & 63;
        vT[(((size_t)(b * HH + h)) * HD + hd) * SS + s] =
            (_Float16)(gate[b] * a + bn);
      }
    }
  }
}

// flash attention: one block = 64 query rows of one (b,h); 4 waves x 16 rows.
// K/V tiles staged once per block via async-to-LDS, shared by all waves.
__global__ __launch_bounds__(128) void attn_kernel(
    const _Float16* __restrict__ qh, const _Float16* __restrict__ kh,
    const _Float16* __restrict__ vth, const float* __restrict__ mask,
    _Float16* __restrict__ aout) {
  __shared__ _Float16 ktile[32][64];              // 32 keys x 64 hd   (4KB)
  __shared__ _Float16 vtile[64][32];              // 64 hd  x 32 keys  (4KB)
  __shared__ _Float16 pstage[4][16][32];          // per-wave P tile   (4KB)

  const int tid  = threadIdx.x;
  const int wave = tid >> 5, lane = tid & 31;
  const int bh = blockIdx.x >> 5;                 // 32 q-blocks per (b,h)
  const int q0 = ((blockIdx.x & 31) << 6) + wave * 16;
  const int b = bh >> 3, h = bh & 7;
  const int half = lane >> 4, nloc = lane & 15;

  const _Float16* qbase = qh + ((size_t)bh * SS + q0) * HD;
  const _Float16* kbh   = kh + (size_t)bh * SS * HD;
  const _Float16* vbh   = vth + (size_t)bh * HD * SS;
  const float*    mbase = mask + ((size_t)h * SS + q0) * SS;

  const v16h a0 = load_a16(qbase, HD);            // q K = 0..31
  const v16h a1 = load_a16(qbase + 32, HD);       // q K = 32..63

  float mi[8], li[8];
#pragma unroll
  for (int r = 0; r < 8; ++r) { mi[r] = -1e30f; li[r] = 0.f; }
  v8f o[4] = {};

  // staging addresses for this thread (16B granules)
  const int vrow = tid >> 1, vcol = (tid & 1) * 8;    // V: row=hd, 8 f16 = 16B

  for (int t0 = 0; t0 < SS; t0 += 32) {
    // ---- cooperative async stage: K tile (contiguous 4KB) + V tile --------
    {
      const char* gk = (const char*)(kbh + (size_t)t0 * HD);
      char* lk = (char*)&ktile[0][0];
      async_b128(lk + tid * 16,        gk + tid * 16);          // bytes 0..2047
      async_b128(lk + tid * 16 + 2048, gk + tid * 16 + 2048);   // bytes 2048..4095
      const _Float16* gv = vbh + (size_t)vrow * SS + t0 + vcol;
      async_b128(&vtile[vrow][vcol],      gv);                  // k = t0..t0+7
      async_b128(&vtile[vrow][vcol + 16], gv + 16);             // k = t0+16..t0+23
    }
    asm volatile("s_wait_asynccnt 0x0" ::: "memory");
    __syncthreads();

    // ---- scores = q(16x64) @ k^T, fragments from LDS ----------------------
    v16h bk00 = load_b16(&ktile[0][0], HD);
    v16h bk01 = load_b16(&ktile[0][32], HD);
    v16h bk10 = load_b16(&ktile[16][0], HD);
    v16h bk11 = load_b16(&ktile[16][32], HD);
    v8f s0 = {}, s1 = {};
    s0 = wmma16(a0, bk00, s0); s0 = wmma16(a1, bk01, s0);
    s1 = wmma16(a0, bk10, s1); s1 = wmma16(a1, bk11, s1);

    if (t0 + 32 < SS)                              // global_prefetch_b8
      __builtin_prefetch(kbh + (size_t)(t0 + 32) * HD, 0, 1);

    // ---- scale + mask (C-layout: reg r -> row r+8*half, col nloc/+16) -----
#pragma unroll
    for (int r = 0; r < 8; ++r) {
      const float* mr = mbase + (size_t)(r + 8 * half) * SS + t0 + nloc;
      s0[r] = s0[r] * 0.125f + mr[0];
      s1[r] = s1[r] * 0.125f + mr[16];
    }

    // ---- online softmax ---------------------------------------------------
    float alpha[8];
#pragma unroll
    for (int r = 0; r < 8; ++r) {
      float mx = row_max16(fmaxf(s0[r], s1[r]));
      float mn = fmaxf(mi[r], mx);
      alpha[r] = __expf(mi[r] - mn);
      mi[r] = mn;
      s0[r] = __expf(s0[r] - mn);
      s1[r] = __expf(s1[r] - mn);
      li[r] = li[r] * alpha[r] + row_sum16(s0[r] + s1[r]);
    }
#pragma unroll
    for (int j = 0; j < 4; ++j)
#pragma unroll
      for (int r = 0; r < 8; ++r) o[j][r] *= alpha[r];

    // ---- P (C-layout f32) -> LDS f16 -> A-layout fragment -----------------
#pragma unroll
    for (int r = 0; r < 8; ++r) {
      const int row = r + 8 * half;
      pstage[wave][row][nloc]      = (_Float16)s0[r];
      pstage[wave][row][nloc + 16] = (_Float16)s1[r];
    }
    asm volatile("s_wait_dscnt 0x0" ::: "memory");
    __builtin_amdgcn_wave_barrier();
    v16h pa = load_a16(&pstage[wave][0][0], 32);

    // ---- O += P(16x32) @ V(32x64), V fragments from LDS -------------------
#pragma unroll
    for (int j = 0; j < 4; ++j) {
      v16h bv = load_b16(&vtile[j * 16][0], 32);
      o[j] = wmma16(pa, bv, o[j]);
    }
    __syncthreads();   // all waves done with ktile/vtile before next overwrite
  }

  float inv[8];
#pragma unroll
  for (int r = 0; r < 8; ++r) inv[r] = 1.0f / li[r];
#pragma unroll
  for (int j = 0; j < 4; ++j) {
#pragma unroll
    for (int r = 0; r < 8; ++r) {
      const int s = q0 + r + 8 * half;
      const int d = h * HD + j * 16 + nloc;
      aout[((size_t)s * BB + b) * DD + d] = (_Float16)(o[j][r] * inv[r]);
    }
  }
}

// out projection: [8192 x 512] @ [512 x 512]^T + bias -> f32
__global__ __launch_bounds__(128) void out_proj_kernel(
    const _Float16* __restrict__ ah, const _Float16* __restrict__ woh,
    const float* __restrict__ bias, float* __restrict__ out) {
  const int wave = threadIdx.x >> 5, lane = threadIdx.x & 31;
  const int m0 = blockIdx.x * 64 + wave * 16;
  const int n0 = blockIdx.y * 64;
  v8f acc[4] = {};
  gemm_16x64(ah + (size_t)m0 * DD, DD, woh + (size_t)n0 * DD, DD, DD, acc);
  const int half = lane >> 4, nloc = lane & 15;
#pragma unroll
  for (int j = 0; j < 4; ++j) {
    const int n = n0 + j * 16 + nloc;
    const float bn = bias[n];
#pragma unroll
    for (int r = 0; r < 8; ++r) {
      const int m = m0 + r + 8 * half;
      out[(size_t)m * DD + n] = acc[j][r] + bn;
    }
  }
}

// ---------------------------------------------------------------------------
extern "C" void kernel_launch(void* const* d_in, const int* in_sizes, int n_in,
                              void* d_out, int out_size, void* d_ws, size_t ws_size,
                              hipStream_t stream) {
  const float* x     = (const float*)d_in[0];   // (S,B,D)
  const float* hmask = (const float*)d_in[1];   // (H,S,S)
  const float* vgate = (const float*)d_in[2];   // (1,B,1)
  const float* in_w  = (const float*)d_in[3];   // (3D,D)
  const float* in_b  = (const float*)d_in[4];   // (3D,)
  const float* out_w = (const float*)d_in[5];   // (D,D)
  const float* out_b = (const float*)d_in[6];   // (D,)
  float* out = (float*)d_out;

  // workspace layout (bytes, 256B aligned); needs >= 44,040,192 B
  char* ws = (char*)d_ws;
  _Float16* xh  = (_Float16*)(ws + 0);          // 8192*512   f16 =  8,388,608
  _Float16* wh  = (_Float16*)(ws + 8388608);    // 1536*512   f16 =  1,572,864
  _Float16* woh = (_Float16*)(ws + 9961472);    //  512*512   f16 =    524,288
  _Float16* qh  = (_Float16*)(ws + 10485760);   // B*H*S*HD   f16 =  8,388,608
  _Float16* kh  = (_Float16*)(ws + 18874368);
  _Float16* vth = (_Float16*)(ws + 27262976);
  _Float16* ah  = (_Float16*)(ws + 35651584);   // 8192*512   f16 =  8,388,608

  // 0) convert fp32 operands to f16
  cvt_f32_to_f16<<<(SS * BB * DD + 255) / 256, 256, 0, stream>>>(x, xh, SS * BB * DD);
  cvt_f32_to_f16<<<(3 * DD * DD + 255) / 256, 256, 0, stream>>>(in_w, wh, 3 * DD * DD);
  cvt_f32_to_f16<<<(DD * DD + 255) / 256, 256, 0, stream>>>(out_w, woh, DD * DD);

  // 1) QKV projection: M=8192, N=1536, K=512
  qkv_proj_kernel<<<dim3(8192 / 64, 1536 / 64), 128, 0, stream>>>(
      xh, wh, in_b, vgate, qh, kh, vth);

  // 2) flash attention: B*H*(S/64) = 1024 blocks, 4 waves each
  attn_kernel<<<1024, 128, 0, stream>>>(qh, kh, vth, hmask, ah);

  // 3) output projection: M=8192, N=512, K=512
  out_proj_kernel<<<dim3(8192 / 64, 512 / 64), 128, 0, stream>>>(ah, woh, out_b, out);
}
